// CTCAligner_84387517432230
// MI455X (gfx1250) — matrix operand order repaired and checked by reading, not compiled, not run on hardware
//
#include <hip/hip_runtime.h>
#include <stdint.h>

typedef unsigned int u32;
typedef unsigned short u16;
typedef __attribute__((ext_vector_type(8)))  float  v8f;
typedef __attribute__((ext_vector_type(16))) __bf16 v16bf;
typedef __attribute__((ext_vector_type(4)))  u32    u32x4;
typedef __attribute__((ext_vector_type(8)))  int    i32x8;
typedef __attribute__((ext_vector_type(4)))  int    i32x4;

#define B_   32
#define T_   2000
#define TM1  1999
#define H_   256
#define S_   63
#define SPAD 64

union AB { v16bf v; u32 u[8]; };

__device__ __forceinline__ u16 f2bf(float x) {
  u32 u = __float_as_uint(x);
  u += 0x7FFFu + ((u >> 16) & 1u);       // round-to-nearest-even
  return (u16)(u >> 16);
}
__device__ __forceinline__ float bf2f(u16 h) { return __uint_as_float(((u32)h) << 16); }

__device__ __forceinline__ v8f wmma_bf16(v16bf a, v16bf b, v8f c) {
  // (neg_a, A, neg_b, B, c_mod, C, reuse_a, reuse_b)
  return __builtin_amdgcn_wmma_f32_16x16x32_bf16(false, a, false, b, (short)0, c, false, false);
}

// ---- Tensor Data Mover: 2D bf16 tile (strided global -> packed LDS) ----
__device__ __forceinline__ void tdm_load_2d_u16(u32 lds_off, const void* gptr,
                                                u32 tensor_d0, u32 tensor_d1,
                                                u32 tile_d0, u32 tile_d1, u32 stride0) {
  unsigned long long ga = (unsigned long long)gptr;
  u32x4 g0; i32x8 g1; i32x4 gz; i32x8 gz8;
  gz[0] = 0; gz[1] = 0; gz[2] = 0; gz[3] = 0;
  gz8[0] = 0; gz8[1] = 0; gz8[2] = 0; gz8[3] = 0;
  gz8[4] = 0; gz8[5] = 0; gz8[6] = 0; gz8[7] = 0;
  g0[0] = 1u;                                   // count=1, user descriptor
  g0[1] = lds_off;                              // LDS byte address
  g0[2] = (u32)ga;                              // global addr lo
  g0[3] = (u32)(ga >> 32) | 0x80000000u;        // global addr hi | type=2 (bits 127:126)
  g1[0] = (int)0x00010000u;                     // data_size=1 (2 bytes), no wg_mask
  g1[1] = (int)((tensor_d0 & 0xFFFFu) << 16);   // tensor_dim0[15:0] at bit 48
  g1[2] = (int)((tensor_d0 >> 16) | ((tensor_d1 & 0xFFFFu) << 16));
  g1[3] = (int)((tensor_d1 >> 16) | (tile_d0 << 16));  // tile_dim0 at bits 127:112
  g1[4] = (int)(tile_d1 & 0xFFFFu);             // tile_dim1, tile_dim2=0
  g1[5] = (int)stride0;                         // tensor_dim0_stride[31:0]
  g1[6] = 0;                                    // stride0 hi16 | stride1 lo16
  g1[7] = 0;                                    // stride1 hi32
  __builtin_amdgcn_tensor_load_to_lds(g0, g1, gz, gz, gz8, 0);
}

// ================= Kernel 0: weight transpose + bf16 hi/lo split =================
__global__ void ctc_prep(const float* __restrict__ W1, const float* __restrict__ W2,
                         const float* __restrict__ b2,
                         u16* __restrict__ w1thi, u16* __restrict__ w1tlo,
                         u16* __restrict__ w2thi, u16* __restrict__ w2tlo,
                         float* __restrict__ b2p) {
  int i = blockIdx.x * 256 + threadIdx.x;
  if (i < 512 * 256) {                          // W1 [512][256] -> W1T [256][512]
    int k = i >> 8, n = i & 255;
    float x = W1[i];
    u16 h = f2bf(x), l = f2bf(x - bf2f(h));
    w1thi[n * 512 + k] = h;
    w1tlo[n * 512 + k] = l;
  } else if (i < 512 * 256 + 256 * 64) {        // W2 [256][63] -> W2T [64][256] (pad)
    int r = i - 512 * 256;
    int n = r >> 6, j = r & 63;
    float x = (j < S_) ? W2[n * S_ + j] : 0.0f;
    u16 h = f2bf(x), l = f2bf(x - bf2f(h));
    w2thi[j * 256 + n] = h;
    w2tlo[j * 256 + n] = l;
  } else if (i < 512 * 256 + 256 * 64 + 64) {
    int j = i - (512 * 256 + 256 * 64);
    b2p[j] = (j < S_) ? b2[j] : 0.0f;
  }
}

// ================= Kernel 1: fused pair-MLP (GEMM1 + ReLU + GEMM2), bf16x3 =================
__global__ __launch_bounds__(256) void ctc_mlp_gemm(
    const float* __restrict__ emb, const float* __restrict__ b1,
    const u16* __restrict__ w1thi, const u16* __restrict__ w1tlo,
    const u16* __restrict__ w2thi, const u16* __restrict__ w2tlo,
    const float* __restrict__ b2p, float* __restrict__ trans) {
  __shared__ u16 sW1hi[256 * 32];               // W1T tile [n=256][k=32]  16KB
  __shared__ u16 sW1lo[256 * 32];               //                         16KB
  __shared__ u16 sW2hi[64 * 256];               // W2T [j=64][n=256]       32KB
  __shared__ u16 sW2lo[64 * 256];               //                         32KB
  __shared__ u16 sHhi[8][16 * 256];             // per-wave h rows         64KB
  __shared__ u16 sHlo[8][16 * 256];             //                         64KB

  const int b    = blockIdx.y;
  const int t0   = blockIdx.x * 128;
  const int wave = threadIdx.x >> 5;
  const int lane = threadIdx.x & 31;
  const int half = lane >> 4;                   // 0: lanes 0-15, 1: lanes 16-31
  const int rowA = lane & 15;

  // cooperative W2T -> LDS (once)
  for (int i = threadIdx.x; i < 64 * 256; i += 256) {
    sW2hi[i] = w2thi[i];
    sW2lo[i] = w2tlo[i];
  }

  const int trow  = t0 + wave * 16 + rowA;      // A-layout row (pair index t)
  const int trowc = trow < (TM1 - 1) ? trow : (TM1 - 1);

  v8f acc[16];
#pragma unroll
  for (int nt = 0; nt < 16; ++nt) acc[nt] = (v8f){0, 0, 0, 0, 0, 0, 0, 0};

  const u32 ldsW1hi = (u32)(uintptr_t)(void*)&sW1hi[0];
  const u32 ldsW1lo = (u32)(uintptr_t)(void*)&sW1lo[0];

  for (int ks = 0; ks < 16; ++ks) {
    __syncthreads();                            // previous tile fully consumed
    if (threadIdx.x < 32) {
      // W1T tile: rows n=0..255, cols k=ks*32..+31 (row stride 512 elems) -> packed [256][32]
      tdm_load_2d_u16(ldsW1hi, w1thi + ks * 32, 512, 256, 32, 256, 512);
      tdm_load_2d_u16(ldsW1lo, w1tlo + ks * 32, 512, 256, 32, 256, 512);
      __builtin_amdgcn_s_wait_tensorcnt(0);
    }
    __syncthreads();

    // A fragment (16x32 of pairs): k<256 -> emb[t+1], k>=256 -> emb[t]
    const float* abase =
        emb + ((size_t)(b * T_ + trowc + (ks < 8 ? 1 : 0))) * H_ + (size_t)(ks & 7) * 32;
    AB ahi, alo;
#pragma unroll
    for (int v = 0; v < 8; ++v) {
      int k = half * 16 + (v << 1);  // K pair base, contiguous k = 16*half + 2v
      float x0 = abase[k], x1 = abase[k + 1];
      u16 h0 = f2bf(x0), h1 = f2bf(x1);
      u16 l0 = f2bf(x0 - bf2f(h0)), l1 = f2bf(x1 - bf2f(h1));
      ahi.u[v] = (u32)h0 | ((u32)h1 << 16);
      alo.u[v] = (u32)l0 | ((u32)l1 << 16);
    }

#pragma unroll
    for (int nt = 0; nt < 16; ++nt) {
      const int n = nt * 16 + rowA;
      const u32* ph = (const u32*)&sW1hi[n * 32];
      const u32* pl = (const u32*)&sW1lo[n * 32];
      AB bhi, blo;
#pragma unroll
      for (int v = 0; v < 8; ++v) {             // K pair dword: half*8 + v
        bhi.u[v] = ph[half * 8 + v];
        blo.u[v] = pl[half * 8 + v];
      }
      acc[nt] = wmma_bf16(ahi.v, bhi.v, acc[nt]);
      acc[nt] = wmma_bf16(alo.v, bhi.v, acc[nt]);
      acc[nt] = wmma_bf16(ahi.v, blo.v, acc[nt]);
    }
  }

  // ---- epilogue GEMM1: +b1, ReLU, bf16 split, stage in LDS (A-layout source for GEMM2) ----
  __syncthreads();
  u16* hh = &sHhi[wave][0];
  u16* hl = &sHlo[wave][0];
#pragma unroll
  for (int nt = 0; nt < 16; ++nt) {
    const int col = nt * 16 + rowA;
    const float bb = b1[col];
#pragma unroll
    for (int v = 0; v < 8; ++v) {
      float x = acc[nt][v] + bb;
      x = fmaxf(x, 0.0f);
      u16 h = f2bf(x), l = f2bf(x - bf2f(h));
      int rowC = v + 8 * half;                  // C-layout row
      hh[rowC * 256 + col] = h;
      hl[rowC * 256 + col] = l;
    }
  }
  __syncthreads();

  // ---- GEMM2: h(16x256) @ W2T -> trans(16x64), bf16x3 ----
  v8f acc2[4];
#pragma unroll
  for (int j = 0; j < 4; ++j) acc2[j] = (v8f){0, 0, 0, 0, 0, 0, 0, 0};

  for (int ks2 = 0; ks2 < 8; ++ks2) {
    AB ahi, alo;
    const u32* pa = (const u32*)&hh[rowA * 256 + ks2 * 32];
    const u32* pb = (const u32*)&hl[rowA * 256 + ks2 * 32];
#pragma unroll
    for (int v = 0; v < 8; ++v) {
      int kd = half * 8 + v;  // A K pair dword, contiguous k order
      ahi.u[v] = pa[kd];
      alo.u[v] = pb[kd];
    }
#pragma unroll
    for (int j = 0; j < 4; ++j) {
      const int col = j * 16 + rowA;
      const u32* qh = (const u32*)&sW2hi[col * 256 + ks2 * 32];
      const u32* ql = (const u32*)&sW2lo[col * 256 + ks2 * 32];
      AB bhi, blo;
#pragma unroll
      for (int v = 0; v < 8; ++v) {
        bhi.u[v] = qh[half * 8 + v];
        blo.u[v] = ql[half * 8 + v];
      }
      acc2[j] = wmma_bf16(ahi.v, bhi.v, acc2[j]);
      acc2[j] = wmma_bf16(alo.v, bhi.v, acc2[j]);
      acc2[j] = wmma_bf16(ahi.v, blo.v, acc2[j]);
    }
  }

  // store trans (padded SPAD=64), C-layout
#pragma unroll
  for (int j = 0; j < 4; ++j) {
    const int col = j * 16 + rowA;
    const float bb = b2p[col];
#pragma unroll
    for (int v = 0; v < 8; ++v) {
      int tt = t0 + wave * 16 + v + 8 * half;
      if (tt < TM1) trans[((size_t)b * TM1 + tt) * SPAD + col] = acc2[j][v] + bb;
    }
  }
}

// ================= Kernel 2: local score assembly + softmax (scan collapsed) =================
__device__ __forceinline__ float wred_max(float v) {
#pragma unroll
  for (int o = 16; o; o >>= 1) v = fmaxf(v, __shfl_xor(v, o, 32));
  return v;
}
__device__ __forceinline__ float wred_sum(float v) {
#pragma unroll
  for (int o = 16; o; o >>= 1) v += __shfl_xor(v, o, 32);
  return v;
}

__global__ __launch_bounds__(256) void ctc_combine(const float* __restrict__ lg,
                                                   const int* __restrict__ kw,
                                                   const float* __restrict__ trans,
                                                   float* __restrict__ out) {
  const int wave = threadIdx.x >> 5, lane = threadIdx.x & 31;
  const int row = blockIdx.x * 8 + wave;        // (b,t) row, 0..63999
  const int b = row / T_, t = row - b * T_;
  const float NEG = -1e30f;
  const float* lgbt = lg + ((size_t)b * T_ + t) * S_;
  const float* trb  = trans + (size_t)b * TM1 * SPAD;
  const int s0 = lane, s1 = lane + 32;

  float sc0 = NEG, sc1 = NEG;
  if (t == 0) {
    // LSE of logits[b,0,:] for f0
    float a0 = (s0 < S_) ? lgbt[s0] : NEG;
    float a1 = (s1 < S_) ? lgbt[s1] : NEG;
    float mx = wred_max(fmaxf(a0, a1));
    float e  = ((s0 < S_) ? __expf(a0 - mx) : 0.0f) + ((s1 < S_) ? __expf(a1 - mx) : 0.0f);
    float lse = mx + __logf(wred_sum(e));
    int tok = kw[b * 32];
    float f0 = lgbt[tok] - lse;
    if (s0 < S_) sc0 = lgbt[S_ + s0] + trb[s0] + (s0 == 0 ? f0 : 0.0f);
    if (s1 < S_) sc1 = lgbt[S_ + s1] + trb[s1];
  } else if (t == TM1) {
    if (s0 < S_) sc0 = lgbt[s0] + trb[(size_t)(TM1 - 1) * SPAD + s0];
    if (s1 < S_) sc1 = lgbt[s1] + trb[(size_t)(TM1 - 1) * SPAD + s1];
  } else {
    const float* tr0 = trb + (size_t)(t - 1) * SPAD;
    const float* tr1 = trb + (size_t)t * SPAD;
    if (s0 < S_) sc0 = lgbt[s0] + lgbt[S_ + s0] + tr0[s0] + tr1[s0];
    if (s1 < S_) sc1 = lgbt[s1] + lgbt[S_ + s1] + tr0[s1] + tr1[s1];
  }

  float mx = wred_max(fmaxf(sc0, sc1));
  float e0 = (s0 < S_) ? __expf(sc0 - mx) : 0.0f;
  float e1 = (s1 < S_) ? __expf(sc1 - mx) : 0.0f;
  float inv = 1.0f / wred_sum(e0 + e1);
  float* o = out + ((size_t)b * T_ + t) * S_;
  if (s0 < S_) o[s0] = e0 * inv;
  if (s1 < S_) o[s1] = e1 * inv;
}

// ================================ launch ================================
extern "C" void kernel_launch(void* const* d_in, const int* in_sizes, int n_in,
                              void* d_out, int out_size, void* d_ws, size_t ws_size,
                              hipStream_t stream) {
  (void)in_sizes; (void)n_in; (void)out_size; (void)ws_size;
  const float* lg  = (const float*)d_in[0];
  const float* emb = (const float*)d_in[1];
  const int*   kw  = (const int*)d_in[2];
  const float* W1  = (const float*)d_in[3];
  const float* b1  = (const float*)d_in[4];
  const float* W2  = (const float*)d_in[5];
  const float* b2  = (const float*)d_in[6];
  float* out = (float*)d_out;

  char* ws = (char*)d_ws;
  float* trans = (float*)ws;
  size_t off = (size_t)B_ * TM1 * SPAD * sizeof(float);
  u16* w1thi = (u16*)(ws + off); off += 512 * 256 * sizeof(u16);
  u16* w1tlo = (u16*)(ws + off); off += 512 * 256 * sizeof(u16);
  u16* w2thi = (u16*)(ws + off); off += 64 * 256 * sizeof(u16);
  u16* w2tlo = (u16*)(ws + off); off += 64 * 256 * sizeof(u16);
  float* b2p = (float*)(ws + off);

  int prepN = 512 * 256 + 256 * 64 + 64;
  ctc_prep<<<(prepN + 255) / 256, 256, 0, stream>>>(W1, W2, b2, w1thi, w1tlo, w2thi, w2tlo, b2p);

  dim3 ggrid(16, B_);  // 16 tiles of 128 pair-rows per batch
  ctc_mlp_gemm<<<ggrid, 256, 0, stream>>>(emb, b1, w1thi, w1tlo, w2thi, w2tlo, b2p, trans);

  ctc_combine<<<(B_ * T_) / 8, 256, 0, stream>>>(lg, kw, trans, out);
}
